// DisentangledSelfAttention_65266323030472
// MI455X (gfx1250) — matrix operand order repaired and checked by reading, not compiled, not run on hardware
//
#include <hip/hip_runtime.h>
#include <hip/hip_bf16.h>
#include <stdint.h>

// ---------- problem constants ----------
#define BB 4
#define LL 256
#define DD 512
#define HH 8
#define RR 64
#define DKK 64

typedef __attribute__((ext_vector_type(16))) _Float16 v16h;
typedef __attribute__((ext_vector_type(4)))  _Float16 v4h;
typedef __attribute__((ext_vector_type(8)))  float    v8f;
typedef __attribute__((ext_vector_type(4)))  float    v4f;

// ---------------------------------------------------------------------------
// f32 -> f16 conversion
// ---------------------------------------------------------------------------
__global__ void cvt_f32_to_f16(const float* __restrict__ src,
                               _Float16* __restrict__ dst, int n) {
  int i = blockIdx.x * blockDim.x + threadIdx.x;
  if (i < n) dst[i] = (_Float16)src[i];
}

// ---------------------------------------------------------------------------
// WMMA GEMM, C[M,N] = A[M,K] * B[N,K]^T (+bias[N]).  A,B f16 row-major.
// One 16x16 C tile per wave, 4 waves per 128-thread block.
// Batched via blockIdx.z split into (zo, zi) with zi = z % innerZ.
// ---------------------------------------------------------------------------
__global__ void gemm_abT_wmma(const _Float16* __restrict__ A,
                              const _Float16* __restrict__ Bm,
                              float* __restrict__ Cf,
                              _Float16* __restrict__ Ch,
                              const float* __restrict__ bias,
                              int M, int N, int K, int lda, int ldb, int ldc,
                              long sAo, long sAi, long sBo, long sBi,
                              long sCo, long sCi, int innerZ) {
  int z = blockIdx.z;
  int zo = z / innerZ, zi = z % innerZ;
  A  += zo * sAo + zi * sAi;
  Bm += zo * sBo + zi * sBi;
  long coff = zo * sCo + zi * sCi;

  int tilesN = N >> 4;
  int totalTiles = (M >> 4) * tilesN;
  int tile = blockIdx.x * 4 + (threadIdx.x >> 5);   // wave-uniform
  if (tile >= totalTiles) return;
  int tM = (tile / tilesN) << 4;
  int tN = (tile % tilesN) << 4;

  int ln  = threadIdx.x & 31;
  int lo  = ln & 15;          // row (A) / col (B,C) within tile
  int hi  = ln >> 4;          // half-wave select

  v8f acc = {};
  for (int k0 = 0; k0 < K; k0 += 32) {
    union { unsigned u[8]; v16h h; } a, b;
    // A fragment: 16x32 f16. VGPR v holds K = (v/4)*16 + hi*8 + (v%4)*2 {,+1}
    const _Float16* Ap = A + (long)(tM + lo) * lda + k0;
#pragma unroll
    for (int v = 0; v < 8; ++v) {
      int kk = ((v >> 2) << 4) + hi * 8 + ((v & 3) << 1);
      a.u[v] = *(const unsigned*)(Ap + kk);
    }
    // B fragment: 32x16 f16, B stored [N,K] row-major (transposed access).
    // VGPR v holds K = (v/4)*16 + hi*8 + (v%4)*2 {,+1}, col N = lo.
    const _Float16* Bp = Bm + (long)(tN + lo) * ldb + k0;
#pragma unroll
    for (int v = 0; v < 8; ++v) b.u[v] = *(const unsigned*)(Bp + ((v >> 2) << 4) + hi * 8 + ((v & 3) << 1));

    acc = __builtin_amdgcn_wmma_f32_16x16x32_f16(false, a.h, false, b.h,
                                                 (short)0, acc, false, false);
  }

  int n = tN + lo;
  float bval = bias ? bias[n] : 0.0f;
#pragma unroll
  for (int i = 0; i < 8; ++i) {
    int m = tM + hi * 8 + i;
    float r = acc[i] + bval;
    long o = coff + (long)m * ldc + n;
    if (Cf) Cf[o] = r;
    if (Ch) Ch[o] = (_Float16)r;
  }
}

// ---------------------------------------------------------------------------
// Per-head transpose of V: Yv[b*256+m][h*64+d] -> Vt[b][h][d][m].
// LDS-tiled (64x64, padded) so both global reads and writes are coalesced.
// Lets the ctx GEMM use the b128-vectorized ABT fragment loader.
// ---------------------------------------------------------------------------
__global__ void transpose_v(const _Float16* __restrict__ Yv,
                            _Float16* __restrict__ Vt) {
  int bh = blockIdx.x;            // b*8 + h
  int b = bh >> 3, h = bh & 7;
  __shared__ _Float16 tile[64][65];
  int t = threadIdx.x;            // 256 threads

  for (int m0 = 0; m0 < LL; m0 += 64) {
    int d = t & 63, r = t >> 6;   // r in 0..3
#pragma unroll
    for (int i = 0; i < 16; ++i) {
      int mm = r * 16 + i;
      tile[mm][d] = Yv[((long)(b * LL + m0 + mm)) * DD + h * DKK + d];
    }
    __syncthreads();
    int mo = t & 63, dg = t >> 6;
#pragma unroll
    for (int i = 0; i < 16; ++i) {
      int dd = dg * 16 + i;
      Vt[((long)bh * DKK + dd) * LL + m0 + mo] = tile[mo][dd];
    }
    __syncthreads();
  }
}

// ---------------------------------------------------------------------------
// Gather (c2p / p2c) + scale + mask + softmax over keys.  One block per
// (b,h,l) row, 256 threads (one per key m).  Writes P in f16 for ctx WMMA.
// ---------------------------------------------------------------------------
__global__ void score_softmax(const float* __restrict__ S,
                              const float* __restrict__ c2p,
                              const float* __restrict__ p2c,
                              const int* __restrict__ relpos,
                              const unsigned char* __restrict__ mask,
                              _Float16* __restrict__ P) {
  int b = blockIdx.z, h = blockIdx.y, l = blockIdx.x, m = threadIdx.x;
  long bh = (long)b * HH + h;

  float s = S[(bh * LL + l) * LL + m];
  int r = relpos[((long)b * LL + l) * LL + m];
  s += c2p[(bh * LL + l) * RR + r] + p2c[(bh * LL + m) * RR + r];
  s *= (1.0f / 24.0f);                       // 1 / (3 * sqrt(64))
  if (mask[b * LL + m]) s = -1e9f;

  __shared__ float red[LL];
  red[m] = s;
  __syncthreads();
  for (int off = LL / 2; off > 0; off >>= 1) {
    if (m < off) red[m] = fmaxf(red[m], red[m + off]);
    __syncthreads();
  }
  float mx = red[0];
  __syncthreads();

  float e = expf(s - mx);
  red[m] = e;
  __syncthreads();
  for (int off = LL / 2; off > 0; off >>= 1) {
    if (m < off) red[m] += red[m + off];
    __syncthreads();
  }
  float inv = 1.0f / red[0];
  P[(bh * LL + l) * LL + m] = (_Float16)(e * inv);
}

// ---------------------------------------------------------------------------
// Streaming rel_v contraction: ctx[b,h,l,d] += sum_m P[b,h,l,m]*rv[b,l,m,h,d]
// This is the HBM roofline of the whole op: 537MB read exactly once.
//  - 2 'l' rows per 256-thread block; each thread owns 4 adjacent columns
//    -> every m-row is a fully coalesced 2KB b128 burst per row.
//  - rel_v loads are NON-TEMPORAL so the zero-reuse 537MB stream does not
//    evict the <30MB of L2-resident intermediates (S, P, Y*, ctx).
// Adds the WMMA ctx (f32) and emits the final f16 ctx for the last GEMM.
// ---------------------------------------------------------------------------
__global__ void relv_stream(const _Float16* __restrict__ P,
                            const float* __restrict__ rv,
                            const float* __restrict__ ctxF,
                            _Float16* __restrict__ ctxH) {
  int b = blockIdx.y, lp = blockIdx.x, t = threadIdx.x;
  int half = t >> 7, tt = t & 127;
  int l = lp * 2 + half;

  __shared__ float pl[2][HH * LL];           // 16KB: probs for both rows
#pragma unroll
  for (int h = 0; h < HH; ++h) {
#pragma unroll
    for (int j = 0; j < 2; ++j) {
      int m = tt + j * 128;
      pl[half][h * LL + m] =
          (float)P[(((long)b * HH + h) * LL + l) * LL + m];
    }
  }
  __syncthreads();

  int c0 = tt * 4;                           // four adjacent columns, one head
  const float* pr = &pl[half][(c0 >> 6) * LL];
  const float* rvp = rv + (((long)b * LL + l) * LL) * DD + c0;

  float a0 = 0.f, a1 = 0.f, a2 = 0.f, a3 = 0.f;
#pragma unroll 4
  for (int m = 0; m < LL; ++m) {
    v4f d = __builtin_nontemporal_load((const v4f*)rvp);
    float p = pr[m];
    a0 = fmaf(p, d.x, a0);
    a1 = fmaf(p, d.y, a1);
    a2 = fmaf(p, d.z, a2);
    a3 = fmaf(p, d.w, a3);
    rvp += DD;
  }

  long row = ((long)b * LL + l) * DD + c0;
  v4f c = *(const v4f*)(ctxF + row);
  v4h o;
  o.x = (_Float16)(c.x + a0);
  o.y = (_Float16)(c.y + a1);
  o.z = (_Float16)(c.z + a2);
  o.w = (_Float16)(c.w + a3);
  *(v4h*)(ctxH + row) = o;
}

// ---------------------------------------------------------------------------
// Host-side orchestration
// ---------------------------------------------------------------------------
extern "C" void kernel_launch(void* const* d_in, const int* in_sizes, int n_in,
                              void* d_out, int out_size, void* d_ws, size_t ws_size,
                              hipStream_t stream) {
  (void)in_sizes; (void)n_in; (void)out_size; (void)ws_size;

  const float* q_in  = (const float*)d_in[0];
  const float* k_in  = (const float*)d_in[1];
  const float* v_in  = (const float*)d_in[2];
  const unsigned char* mask = (const unsigned char*)d_in[3];
  const int*   relpos = (const int*)d_in[4];
  const float* rel_q = (const float*)d_in[5];
  const float* rel_k = (const float*)d_in[6];
  const float* rel_v = (const float*)d_in[7];
  const float* Wq = (const float*)d_in[8];   const float* bq = (const float*)d_in[9];
  const float* Wk = (const float*)d_in[10];  const float* bk = (const float*)d_in[11];
  const float* Wv = (const float*)d_in[12];  const float* bv = (const float*)d_in[13];
  const float* Wo = (const float*)d_in[14];  const float* bo = (const float*)d_in[15];
  float* out = (float*)d_out;

  const int NTOK = BB * LL;                  // 1024 rows
  const long nX   = (long)NTOK * DD;         // 524288
  const long nW   = (long)DD * DD;           // 262144
  const long nRel = (long)HH * RR * DKK;     // 32768
  const long nS   = (long)BB * HH * LL * LL; // 2097152
  const long nCP  = (long)BB * HH * LL * RR; // 524288

  // workspace carve-up (all chunks 256B aligned by construction)
  char* w = (char*)d_ws;
  size_t off = 0;
  auto take = [&](size_t bytes) { char* p = w + off; off += (bytes + 255) & ~(size_t)255; return p; };
  _Float16* Xq16  = (_Float16*)take(nX * 2);
  _Float16* Xk16  = (_Float16*)take(nX * 2);
  _Float16* Xv16  = (_Float16*)take(nX * 2);
  _Float16* Wq16  = (_Float16*)take(nW * 2);
  _Float16* Wk16  = (_Float16*)take(nW * 2);
  _Float16* Wv16  = (_Float16*)take(nW * 2);
  _Float16* Wo16  = (_Float16*)take(nW * 2);
  _Float16* rq16  = (_Float16*)take(nRel * 2);
  _Float16* rk16  = (_Float16*)take(nRel * 2);
  _Float16* Yq16  = (_Float16*)take(nX * 2);
  _Float16* Yk16  = (_Float16*)take(nX * 2);
  _Float16* Yv16  = (_Float16*)take(nX * 2);
  _Float16* Vt16  = (_Float16*)take(nX * 2);   // V^T per head: [b][h][d][m]
  float*    S     = (float*)   take(nS * 4);
  float*    c2p   = (float*)   take(nCP * 4);
  float*    p2c   = (float*)   take(nCP * 4);
  _Float16* P16   = (_Float16*)take(nS * 2);
  float*    ctxF  = (float*)   take(nX * 4);
  _Float16* ctxH  = (_Float16*)take(nX * 2);

  // ---- stage 0: f32 -> f16 converts ----
  auto cvt = [&](const float* s, _Float16* d, long n) {
    cvt_f32_to_f16<<<dim3((unsigned)((n + 255) / 256)), dim3(256), 0, stream>>>(s, d, (int)n);
  };
  cvt(q_in, Xq16, nX);  cvt(k_in, Xk16, nX);  cvt(v_in, Xv16, nX);
  cvt(Wq, Wq16, nW);    cvt(Wk, Wk16, nW);    cvt(Wv, Wv16, nW);  cvt(Wo, Wo16, nW);
  cvt(rel_q, rq16, nRel);  cvt(rel_k, rk16, nRel);

  // ---- stage 1: QKV projections  Y = X * W^T + b  (1024x512x512) ----
  dim3 blk(128);
  {
    dim3 grd(512, 1, 1);  // 2048 tiles / 4 waves
    gemm_abT_wmma<<<grd, blk, 0, stream>>>(Xq16, Wq16, nullptr, Yq16, bq,
        NTOK, DD, DD, DD, DD, DD, 0, 0, 0, 0, 0, 0, 1);
    gemm_abT_wmma<<<grd, blk, 0, stream>>>(Xk16, Wk16, nullptr, Yk16, bk,
        NTOK, DD, DD, DD, DD, DD, 0, 0, 0, 0, 0, 0, 1);
    gemm_abT_wmma<<<grd, blk, 0, stream>>>(Xv16, Wv16, nullptr, Yv16, bv,
        NTOK, DD, DD, DD, DD, DD, 0, 0, 0, 0, 0, 0, 1);
  }

  // ---- stage 1b: per-head transpose of V for the ctx GEMM ----
  transpose_v<<<dim3(BB * HH), dim3(256), 0, stream>>>(Yv16, Vt16);

  // ---- stage 2: scores S[b,h] = q_bh (256x64) * k_bh^T ----
  {
    dim3 grd(64, 1, BB * HH);  // 256 tiles / 4 waves, 32 batches
    gemm_abT_wmma<<<grd, blk, 0, stream>>>(Yq16, Yk16, S, nullptr, nullptr,
        LL, LL, DKK, DD, DD, LL,
        (long)LL * DD, (long)DKK,          // A: +b*256*512, +h*64
        (long)LL * DD, (long)DKK,          // B: same layout
        (long)HH * LL * LL, (long)LL * LL, // C: [b][h][l][m]
        HH);
  }
  // ---- stage 2b: c2p = q * rel_k^T,  p2c = k * rel_q^T  (256x64x64) ----
  {
    dim3 grd(16, 1, BB * HH);
    gemm_abT_wmma<<<grd, blk, 0, stream>>>(Yq16, rk16, c2p, nullptr, nullptr,
        LL, RR, DKK, DD, DKK, RR,
        (long)LL * DD, (long)DKK,
        0L, (long)RR * DKK,
        (long)HH * LL * RR, (long)LL * RR,
        HH);
    gemm_abT_wmma<<<grd, blk, 0, stream>>>(Yk16, rq16, p2c, nullptr, nullptr,
        LL, RR, DKK, DD, DKK, RR,
        (long)LL * DD, (long)DKK,
        0L, (long)RR * DKK,
        (long)HH * LL * RR, (long)LL * RR,
        HH);
  }

  // ---- stage 3: gather + scale + mask + softmax -> P (f16) ----
  {
    dim3 grd(LL, HH, BB);
    score_softmax<<<grd, dim3(LL), 0, stream>>>(S, c2p, p2c, relpos, mask, P16);
  }

  // ---- stage 4: ctx = P * Vt^T  (256x64, K=256 per (b,h)) ----
  {
    dim3 grd(16, 1, BB * HH);  // 64 tiles / 4 waves
    gemm_abT_wmma<<<grd, blk, 0, stream>>>(P16, Vt16, ctxF, nullptr, nullptr,
        LL, DKK, LL, LL, LL, DD,
        (long)HH * LL * LL, (long)LL * LL,  // A = P [b][h][l][m]
        (long)HH * DKK * LL, (long)DKK * LL,// B = Vt [b][h][d][m]
        (long)LL * DD, (long)DKK,           // C = ctx rows b*256+l, cols h*64+d
        HH);
  }

  // ---- stage 5: stream rel_v (537MB NT, the HBM roofline), add, emit f16 --
  {
    dim3 grd(LL / 2, BB);
    relv_stream<<<grd, dim3(256), 0, stream>>>(P16, rel_v, ctxF, ctxH);
  }

  // ---- stage 6: out = ctx * Wo^T + bo ----
  {
    dim3 grd(512, 1, 1);
    gemm_abT_wmma<<<grd, blk, 0, stream>>>(ctxH, Wo16, out, nullptr, bo,
        NTOK, DD, DD, DD, DD, DD, 0, 0, 0, 0, 0, 0, 1);
  }
}
